// Vulnerability_20727512170905
// MI455X (gfx1250) — compile-verified, matrix operand order checked by reading, not run on hardware
//
#include <hip/hip_runtime.h>
#include <hip/hip_bf16.h>

// ---------------------------------------------------------------------------
// Types for CDNA5 WMMA (wave32, 16x16x32 bf16 -> f32)
// ---------------------------------------------------------------------------
typedef __attribute__((ext_vector_type(16))) __bf16 v16bf;
typedef __attribute__((ext_vector_type(8)))  __bf16 v8bf;
typedef __attribute__((ext_vector_type(8)))  float  v8f;

#define TPB 256
#define NROWS 16384
#define DD 128
#define HH 128
#define TT 15
#define LL 16
#define EE 262144

__device__ inline float sigmf_(float x) { return 1.0f / (1.0f + __expf(-x)); }

// ---------------------------------------------------------------------------
// Pack W[K x Nc] (f32, row-major) into per-fragment bf16 layout:
//   Wp[((nt*KC + kc)*32 + lane)*16 + j]
// per CDNA5 ISA 7.12.2 (16-bit B fragment, symmetric to A):
//   col  = nt*16 + (lane&15), half = lane>>4
//   j<8 : K = kc*32 + 8*half + j ;  j>=8: K = kc*32 + 16 + 8*half + (j-8)
// Each lane's fragment is 32 contiguous bytes -> two global_load_b128.
// ---------------------------------------------------------------------------
__global__ void k_pack_b(const float* __restrict__ W, __hip_bfloat16* __restrict__ Wp,
                         int K, int Nc) {
    int idx = blockIdx.x * TPB + threadIdx.x;
    if (idx >= K * Nc) return;
    int j    = idx & 15;
    int lane = (idx >> 4) & 31;
    int rest = idx >> 9;
    int KC   = K >> 5;
    int kc   = rest % KC;
    int nt   = rest / KC;
    int half = lane >> 4;
    int col  = nt * 16 + (lane & 15);
    int k    = kc * 32 + 8 * half + ((j < 8) ? j : (8 + j));   // +16 for upper half
    Wp[idx] = __float2bfloat16(W[(size_t)k * Nc + col]);
}

__global__ void k_add_vec(const float* a, const float* b, float* o, int n) {
    int i = blockIdx.x * TPB + threadIdx.x;
    if (i < n) o[i] = a[i] + b[i];
}

__global__ void k_zero(float* p, int n) {
    int i = blockIdx.x * TPB + threadIdx.x;
    if (i < n) p[i] = 0.0f;
}

// ---------------------------------------------------------------------------
// WMMA GEMM:  C[M x Nc] = A[M x K](f32, lda) * Wp(packed bf16) (+bias) (+Cin)
// Block (8 waves) computes a 64-row x 128-col tile. The 64x32 f32 A chunk is
// cooperatively converted to bf16 and staged in LDS in fragment order ONCE,
// then all 8 waves (8 column tiles) consume it via ds_load_b128. Double
// buffered; B fragments come pre-packed from global (32B/lane, L0-shared).
// A fragments are preloaded before the WMMA burst so LDS latency overlaps
// with matrix-unit execution.
// ---------------------------------------------------------------------------
__global__ __launch_bounds__(TPB)
void k_gemm_wmma(const float* __restrict__ A, long lda,
                 const __hip_bfloat16* __restrict__ Wp,
                 const float* __restrict__ bias,
                 const float* Cin, long ldcin,
                 float* C, long ldc,
                 int K, int numMt) {
    __shared__ v16bf smA[2][4][32];                 // 8 KB, fragment order
    const v16bf* Wv = reinterpret_cast<const v16bf*>(Wp);
    int tid  = threadIdx.x;
    int wave = tid >> 5;
    int lane = tid & 31;
    int mt   = blockIdx.x % numMt;
    int ntg  = blockIdx.x / numMt;
    int nt   = ntg * 8 + wave;                      // one column tile per wave
    int l16  = lane & 15;
    int half = lane >> 4;
    int col  = nt * 16 + l16;
    int KC   = K >> 5;

    // cooperative staging decomposition: 2048 bf16 per chunk = 8 per thread
    int sj0   = (tid & 1) * 8;
    int slane = (tid >> 1) & 31;
    int sm    = tid >> 6;
    int shalf = slane >> 4;
    long srow = (long)mt * 64 + sm * 16 + (slane & 15);
    const float* Abase = A + srow * lda + 8 * shalf + (sj0 ? 16 : 0);
    __bf16* sd0 = (__bf16*)&smA[0][sm][slane] + sj0;
    __bf16* sd1 = (__bf16*)&smA[1][sm][slane] + sj0;

    auto stage = [&](int buf, int kc) {
        const float* src = Abase + (size_t)kc * 32;
        v8bf v;
#pragma unroll
        for (int j = 0; j < 8; ++j) v[j] = (__bf16)src[j];
        *(v8bf*)(buf ? sd1 : sd0) = v;
    };

    v8f acc[4];
#pragma unroll
    for (int m = 0; m < 4; ++m)
#pragma unroll
        for (int r = 0; r < 8; ++r) acc[m][r] = 0.0f;

    const v16bf* bp = Wv + (size_t)nt * KC * 32 + lane;
    stage(0, 0);
    for (int kc = 0; kc < KC; ++kc) {
        __syncthreads();                            // staged chunk kc visible
        if (kc + 1 < KC) {
            stage((kc + 1) & 1, kc + 1);            // overlap next stage
            __builtin_prefetch(bp + (size_t)(kc + 1) * 32, 0, 3);
        }
        v16bf b = bp[(size_t)kc * 32];              // 32B contiguous per lane
        int cur = kc & 1;
        // preload all 4 A fragments (8 pipelined ds_load_b128), then 4 WMMAs
        v16bf a0 = smA[cur][0][lane];
        v16bf a1 = smA[cur][1][lane];
        v16bf a2 = smA[cur][2][lane];
        v16bf a3 = smA[cur][3][lane];
        acc[0] = __builtin_amdgcn_wmma_f32_16x16x32_bf16(
            false, a0, false, b, (short)0, acc[0], false, false);
        acc[1] = __builtin_amdgcn_wmma_f32_16x16x32_bf16(
            false, a1, false, b, (short)0, acc[1], false, false);
        acc[2] = __builtin_amdgcn_wmma_f32_16x16x32_bf16(
            false, a2, false, b, (short)0, acc[2], false, false);
        acc[3] = __builtin_amdgcn_wmma_f32_16x16x32_bf16(
            false, a3, false, b, (short)0, acc[3], false, false);
    }

    float bv = bias ? bias[col] : 0.0f;
#pragma unroll
    for (int m = 0; m < 4; ++m) {
#pragma unroll
        for (int r = 0; r < 8; ++r) {
            long row = (long)mt * 64 + m * 16 + r + 8 * half;
            float v = acc[m][r] + bv;
            if (Cin) v += Cin[row * ldcin + col];
            C[row * ldc + col] = v;
        }
    }
}

// ---------------------------------------------------------------------------
// Tree-LSTM cell
// ---------------------------------------------------------------------------
__global__ void k_tree_cell(const float* __restrict__ iou,
                            const float* f0, const float* f1, const float* fx,
                            const float* c0, const float* c1,
                            float* __restrict__ h, float* __restrict__ c,
                            int n, int hasChild) {
    int idx = blockIdx.x * TPB + threadIdx.x;
    if (idx >= n) return;
    int i = idx >> 7, j = idx & 127;
    const float* r = iou + (size_t)i * 384;
    float iv = sigmf_(r[j]);
    float ov = sigmf_(r[128 + j]);
    float uv = tanhf(r[256 + j]);
    float cv = iv * uv;
    if (hasChild) {
        float fxv = fx[idx];
        cv += sigmf_(f0[idx] + fxv) * c0[idx];
        cv += sigmf_(f1[idx] + fxv) * c1[idx];
    }
    c[idx] = cv;
    h[idx] = ov * tanhf(cv);
}

// ---------------------------------------------------------------------------
// GRU cell: h_out = (1-z)*n + z*h_prev   (strided h pointers)
// ---------------------------------------------------------------------------
__global__ void k_gru_cell(const float* __restrict__ gi, const float* __restrict__ gh,
                           const float* __restrict__ hprev, int sp,
                           float* __restrict__ hout, int so, int n) {
    int idx = blockIdx.x * TPB + threadIdx.x;
    if (idx >= n) return;
    int i = idx >> 7, j = idx & 127;
    const float* gir = gi + (size_t)i * 384;
    const float* ghr = gh + (size_t)i * 384;
    float r  = sigmf_(gir[j] + ghr[j]);
    float z  = sigmf_(gir[128 + j] + ghr[128 + j]);
    float nn = tanhf(gir[256 + j] + r * ghr[256 + j]);
    float hp = hprev[(size_t)i * sp + j];
    hout[(size_t)i * so + j] = (1.0f - z) * nn + z * hp;
}

__global__ void k_copy_strided(const float* __restrict__ src, int ss,
                               float* __restrict__ dst, int ds, int n, int cols) {
    int idx = blockIdx.x * TPB + threadIdx.x;
    if (idx >= n) return;
    int i = idx / cols, j = idx % cols;
    dst[(size_t)i * ds + j] = src[(size_t)i * ss + j];
}

// fv[i, t*256 + r] = r<128 ? fwd[i,t,r] : bwd[i,t,r-128]
__global__ void k_interleave(const float* __restrict__ fwd, const float* __restrict__ bwd,
                             float* __restrict__ fv, int n) {
    int idx = blockIdx.x * TPB + threadIdx.x;
    if (idx >= n) return;
    int i = idx / 1280, k = idx % 1280;
    int t = k >> 8, r = k & 255;
    float v = (r < 128) ? fwd[(size_t)i * 640 + t * 128 + r]
                        : bwd[(size_t)i * 640 + t * 128 + (r - 128)];
    fv[idx] = v;
}

// ---------------------------------------------------------------------------
// GCN helpers
// ---------------------------------------------------------------------------
__global__ void k_deg_init(float* deg, int n) {
    int i = blockIdx.x * TPB + threadIdx.x;
    if (i < n) deg[i] = 1.0f;                     // self loop
}
__global__ void k_deg_accum(const int* __restrict__ dst, float* deg, int e) {
    int i = blockIdx.x * TPB + threadIdx.x;
    if (i < e) atomicAdd(&deg[dst[i]], 1.0f);
}
__global__ void k_rsqrt(const float* deg, float* dinv, int n) {
    int i = blockIdx.x * TPB + threadIdx.x;
    if (i < n) dinv[i] = rsqrtf(deg[i]);
}
__global__ void k_bias_init(const float* __restrict__ b, float* o, int n, int cols) {
    int i = blockIdx.x * TPB + threadIdx.x;
    if (i < n) o[i] = b[i % cols];
}
__global__ void k_selfloop(const float* __restrict__ xw, const float* __restrict__ dinv,
                           float* o, int n, int cols) {
    int idx = blockIdx.x * TPB + threadIdx.x;
    if (idx >= n) return;
    int i = idx / cols;
    float dv = dinv[i];
    o[idx] += dv * dv * xw[idx];
}
__global__ void k_edge_scatter(const int* __restrict__ src, const int* __restrict__ dst,
                               const float* __restrict__ dinv, const float* __restrict__ xw,
                               float* o, int e, int cols) {
    int idx = blockIdx.x * TPB + threadIdx.x;
    if (idx >= e * cols) return;
    int ei = idx / cols, j = idx % cols;
    int s = src[ei], d = dst[ei];
    atomicAdd(&o[(size_t)d * cols + j], dinv[s] * dinv[d] * xw[(size_t)s * cols + j]);
}
__global__ void k_relu(float* x, int n) {
    int i = blockIdx.x * TPB + threadIdx.x;
    if (i < n) x[i] = fmaxf(x[i], 0.0f);
}
// tiny GEMM for the Nc=5 classifier head
__global__ void k_small_gemm(const float* __restrict__ X, const float* __restrict__ W,
                             float* o, int n, int K, int Nc) {
    int idx = blockIdx.x * TPB + threadIdx.x;
    if (idx >= n * Nc) return;
    int i = idx / Nc, j = idx % Nc;
    float acc = 0.0f;
    const float* xr = X + (size_t)i * K;
    for (int k = 0; k < K; ++k) acc += xr[k] * W[(size_t)k * Nc + j];
    o[idx] = acc;
}

// ---------------------------------------------------------------------------
// Max-pool over rows (N x 5) + softmax
// ---------------------------------------------------------------------------
__global__ void k_rowmax_partial(const float* __restrict__ x, float* partial, int n) {
    __shared__ float s[TPB];
    int i = blockIdx.x * TPB + threadIdx.x;
    float v[5];
    for (int c = 0; c < 5; ++c) v[c] = (i < n) ? x[(size_t)i * 5 + c] : -3.0e38f;
    for (int c = 0; c < 5; ++c) {
        s[threadIdx.x] = v[c];
        __syncthreads();
        for (int st = TPB / 2; st > 0; st >>= 1) {
            if (threadIdx.x < st) s[threadIdx.x] = fmaxf(s[threadIdx.x], s[threadIdx.x + st]);
            __syncthreads();
        }
        if (threadIdx.x == 0) partial[blockIdx.x * 5 + c] = s[0];
        __syncthreads();
    }
}
__global__ void k_softmax_final(const float* __restrict__ partial, float* out, int nblk) {
    if (threadIdx.x != 0 || blockIdx.x != 0) return;
    float m[5];
    for (int c = 0; c < 5; ++c) m[c] = -3.0e38f;
    for (int b = 0; b < nblk; ++b)
        for (int c = 0; c < 5; ++c) m[c] = fmaxf(m[c], partial[b * 5 + c]);
    float mx = m[0];
    for (int c = 1; c < 5; ++c) mx = fmaxf(mx, m[c]);
    float sum = 0.0f, e[5];
    for (int c = 0; c < 5; ++c) { e[c] = __expf(m[c] - mx); sum += e[c]; }
    for (int c = 0; c < 5; ++c) out[c] = e[c] / sum;
}

// ---------------------------------------------------------------------------
// Host side
// ---------------------------------------------------------------------------
static inline int cdiv(long a, int b) { return (int)((a + b - 1) / b); }

static void gemm(const float* A, long lda, const __hip_bfloat16* Wp,
                 const float* bias, const float* Cin, long ldcin,
                 float* C, long ldc, int K, int Nc, hipStream_t s) {
    int numMt = NROWS / 64;
    int numNtG = Nc / 128;                  // 8 column tiles (8 waves) per block
    int blocks = numMt * numNtG;
    k_gemm_wmma<<<blocks, TPB, 0, s>>>(A, lda, Wp, bias, Cin, ldcin, C, ldc, K, numMt);
}

extern "C" void kernel_launch(void* const* d_in, const int* in_sizes, int n_in,
                              void* d_out, int out_size, void* d_ws, size_t ws_size,
                              hipStream_t stream) {
    (void)in_sizes; (void)n_in; (void)out_size; (void)ws_size;
    const int N = NROWS, H = HH, E = EE;
    const float* tree_x = (const float*)d_in[0];
    const float* seqs[4] = {(const float*)d_in[1], (const float*)d_in[2],
                            (const float*)d_in[3], (const float*)d_in[4]};
    const int* edge = (const int*)d_in[6];
    const int* esrc = edge;
    const int* edst = edge + E;

    const float* tW_ioux = (const float*)d_in[7];  const float* tb_ioux = (const float*)d_in[8];
    const float* tW_iouh = (const float*)d_in[9];  const float* tb_iouh = (const float*)d_in[10];
    const float* tW_fx   = (const float*)d_in[11]; const float* tb_fx   = (const float*)d_in[12];
    const float* tW_fh   = (const float*)d_in[13]; const float* tb_fh   = (const float*)d_in[14];
    const float* gWih[4], *gbih[4], *gWhh[4], *gbhh[4];
    for (int g = 0; g < 4; ++g) {
        gWih[g] = (const float*)d_in[15 + 4 * g]; gbih[g] = (const float*)d_in[16 + 4 * g];
        gWhh[g] = (const float*)d_in[17 + 4 * g]; gbhh[g] = (const float*)d_in[18 + 4 * g];
    }
    const float* cfWih = (const float*)d_in[31]; const float* cfbih = (const float*)d_in[32];
    const float* cfWhh = (const float*)d_in[33]; const float* cfbhh = (const float*)d_in[34];
    const float* cbWih = (const float*)d_in[35]; const float* cbbih = (const float*)d_in[36];
    const float* cbWhh = (const float*)d_in[37]; const float* cbbhh = (const float*)d_in[38];
    const float* connW = (const float*)d_in[39]; const float* connb = (const float*)d_in[40];
    const float* c0W   = (const float*)d_in[41]; const float* c0b   = (const float*)d_in[42];
    const float* c1W   = (const float*)d_in[43]; const float* c1b   = (const float*)d_in[44];
    const float* c2W   = (const float*)d_in[45]; const float* c2b   = (const float*)d_in[46];

    // ---- workspace bump allocator ----
    char* ws = (char*)d_ws;
    size_t off = 0;
    auto allocF = [&](size_t n) { float* p = (float*)(ws + off); off = (off + n * 4 + 255) & ~(size_t)255; return p; };
    auto allocB = [&](size_t n) { __hip_bfloat16* p = (__hip_bfloat16*)(ws + off); off = (off + n * 2 + 255) & ~(size_t)255; return p; };
    auto pack = [&](const float* src, int K, int Nc) {
        __hip_bfloat16* d = allocB((size_t)K * Nc);
        k_pack_b<<<cdiv((long)K * Nc, TPB), TPB, 0, stream>>>(src, d, K, Nc);
        return d;
    };

    // pre-packed bf16 weights (WMMA fragment order)
    __hip_bfloat16* bW_ioux = pack(tW_ioux, 128, 384);
    __hip_bfloat16* bW_iouh = pack(tW_iouh, 128, 384);
    __hip_bfloat16* bW_fx   = pack(tW_fx,   128, 128);
    __hip_bfloat16* bW_fh   = pack(tW_fh,   128, 128);
    __hip_bfloat16 *bgWih[4], *bgWhh[4];
    for (int g = 0; g < 4; ++g) { bgWih[g] = pack(gWih[g], 128, 384); bgWhh[g] = pack(gWhh[g], 128, 384); }
    __hip_bfloat16* bcfWih = pack(cfWih, 128, 384);
    __hip_bfloat16* bcfWhh = pack(cfWhh, 128, 384);
    __hip_bfloat16* bcbWih = pack(cbWih, 128, 384);
    __hip_bfloat16* bcbWhh = pack(cbWhh, 128, 384);
    __hip_bfloat16* bconnW = pack(connW, 1280, 128);
    __hip_bfloat16* bc0W   = pack(c0W,   128, 128);
    __hip_bfloat16* bc1W   = pack(c1W,   128, 128);

    float* comb_bias = allocF(384);
    k_add_vec<<<2, TPB, 0, stream>>>(tb_ioux, tb_iouh, comb_bias, 384);

    // activation buffers
    size_t NH = (size_t)N * H;
    float* hs   = allocF((size_t)TT * NH);
    float* cs   = allocF((size_t)TT * NH);
    float* iou  = allocF((size_t)N * 384);
    float* hsum = allocF(NH);
    float* fxb  = allocF(NH);
    float* f0b  = allocF(NH);
    float* f1b  = allocF(NH);
    float* ghb  = allocF((size_t)N * 384);
    float* hbuf = allocF(NH);
    float* zbuf = allocF(NH);
    float* feat = allocF((size_t)N * 640);
    float* fwd  = allocF((size_t)N * 640);
    float* bwd  = allocF((size_t)N * 640);
    float* fv   = allocF((size_t)N * 1280);
    float* x0   = allocF(NH);
    float* xw   = allocF(NH);
    float* x1   = allocF(NH);
    float* degb = allocF(N);
    float* dinv = allocF(N);
    float* xw5  = allocF((size_t)N * 5);
    float* out5 = allocF((size_t)N * 5);
    float* part = allocF(64 * 5);

    int gNH = cdiv((long)NH, TPB);

    // ---- Tree-LSTM ----
    static const int ch0[TT] = {-1,-1,-1,-1,-1,-1,-1,-1, 0, 2, 4, 6, 8, 10, 12};
    static const int ch1[TT] = {-1,-1,-1,-1,-1,-1,-1,-1, 1, 3, 5, 7, 9, 11, 13};
    for (int t = 0; t < TT; ++t) {
        const float* xs = tree_x + (size_t)t * DD;     // row stride T*D
        gemm(xs, (long)TT * DD, bW_ioux, comb_bias, nullptr, 0, iou, 384, 128, 384, stream);
        int hasChild = (ch0[t] >= 0);
        const float *pc0 = nullptr, *pc1 = nullptr;
        if (hasChild) {
            float* hc0 = hs + (size_t)ch0[t] * NH;
            float* hc1 = hs + (size_t)ch1[t] * NH;
            pc0 = cs + (size_t)ch0[t] * NH;
            pc1 = cs + (size_t)ch1[t] * NH;
            k_add_vec<<<gNH, TPB, 0, stream>>>(hc0, hc1, hsum, (int)NH);
            gemm(hsum, 128, bW_iouh, nullptr, iou, 384, iou, 384, 128, 384, stream);
            gemm(xs, (long)TT * DD, bW_fx, tb_fx, nullptr, 0, fxb, 128, 128, 128, stream);
            gemm(hc0, 128, bW_fh, tb_fh, nullptr, 0, f0b, 128, 128, 128, stream);
            gemm(hc1, 128, bW_fh, tb_fh, nullptr, 0, f1b, 128, 128, 128, stream);
        }
        k_tree_cell<<<gNH, TPB, 0, stream>>>(iou, f0b, f1b, fxb, pc0, pc1,
                                             hs + (size_t)t * NH, cs + (size_t)t * NH,
                                             (int)NH, hasChild);
    }
    // feat slot 0 = tree final hidden
    k_copy_strided<<<gNH, TPB, 0, stream>>>(hs + (size_t)(TT - 1) * NH, 128, feat + 0, 640, (int)NH, 128);

    // ---- four sequence GRUs ----
    for (int g = 0; g < 4; ++g) {
        k_zero<<<gNH, TPB, 0, stream>>>(hbuf, (int)NH);
        for (int t = 0; t < LL; ++t) {
            gemm(seqs[g] + (size_t)t * DD, (long)LL * DD, bgWih[g], gbih[g], nullptr, 0, iou, 384, 128, 384, stream);
            gemm(hbuf, 128, bgWhh[g], gbhh[g], nullptr, 0, ghb, 384, 128, 384, stream);
            k_gru_cell<<<gNH, TPB, 0, stream>>>(iou, ghb, hbuf, 128, hbuf, 128, (int)NH);
        }
        k_copy_strided<<<gNH, TPB, 0, stream>>>(hbuf, 128, feat + (size_t)(g + 1) * 128, 640, (int)NH, 128);
    }

    // ---- bidirectional GRU over the 5 features ----
    k_zero<<<gNH, TPB, 0, stream>>>(zbuf, (int)NH);
    for (int t = 0; t < 5; ++t) {                       // forward
        const float* hp = (t == 0) ? zbuf : fwd + (size_t)(t - 1) * 128;
        int sp = (t == 0) ? 128 : 640;
        gemm(feat + (size_t)t * 128, 640, bcfWih, cfbih, nullptr, 0, iou, 384, 128, 384, stream);
        gemm(hp, sp, bcfWhh, cfbhh, nullptr, 0, ghb, 384, 128, 384, stream);
        k_gru_cell<<<gNH, TPB, 0, stream>>>(iou, ghb, hp, sp, fwd + (size_t)t * 128, 640, (int)NH);
    }
    for (int t = 4; t >= 0; --t) {                      // backward
        const float* hp = (t == 4) ? zbuf : bwd + (size_t)(t + 1) * 128;
        int sp = (t == 4) ? 128 : 640;
        gemm(feat + (size_t)t * 128, 640, bcbWih, cbbih, nullptr, 0, iou, 384, 128, 384, stream);
        gemm(hp, sp, bcbWhh, cbbhh, nullptr, 0, ghb, 384, 128, 384, stream);
        k_gru_cell<<<gNH, TPB, 0, stream>>>(iou, ghb, hp, sp, bwd + (size_t)t * 128, 640, (int)NH);
    }

    // ---- concat + conn linear ----
    k_interleave<<<cdiv((long)N * 1280, TPB), TPB, 0, stream>>>(fwd, bwd, fv, N * 1280);
    gemm(fv, 1280, bconnW, connb, nullptr, 0, x0, 128, 1280, 128, stream);

    // ---- GCN normalization ----
    k_deg_init<<<cdiv(N, TPB), TPB, 0, stream>>>(degb, N);
    k_deg_accum<<<cdiv(E, TPB), TPB, 0, stream>>>(edst, degb, E);
    k_rsqrt<<<cdiv(N, TPB), TPB, 0, stream>>>(degb, dinv, N);

    // layer 0: x1 = relu(agg(x0 @ c0W) + c0b)
    gemm(x0, 128, bc0W, nullptr, nullptr, 0, xw, 128, 128, 128, stream);
    k_bias_init<<<gNH, TPB, 0, stream>>>(c0b, x1, (int)NH, 128);
    k_selfloop<<<gNH, TPB, 0, stream>>>(xw, dinv, x1, (int)NH, 128);
    k_edge_scatter<<<cdiv((long)E * 128, TPB), TPB, 0, stream>>>(esrc, edst, dinv, xw, x1, E, 128);
    k_relu<<<gNH, TPB, 0, stream>>>(x1, (int)NH);

    // layer 1: x0 = relu(agg(x1 @ c1W) + c1b)
    gemm(x1, 128, bc1W, nullptr, nullptr, 0, xw, 128, 128, 128, stream);
    k_bias_init<<<gNH, TPB, 0, stream>>>(c1b, x0, (int)NH, 128);
    k_selfloop<<<gNH, TPB, 0, stream>>>(xw, dinv, x0, (int)NH, 128);
    k_edge_scatter<<<cdiv((long)E * 128, TPB), TPB, 0, stream>>>(esrc, edst, dinv, xw, x0, E, 128);
    k_relu<<<gNH, TPB, 0, stream>>>(x0, (int)NH);

    // layer 2 (Nc=5): out5 = agg(x0 @ c2W) + c2b
    k_small_gemm<<<cdiv((long)N * 5, TPB), TPB, 0, stream>>>(x0, c2W, xw5, N, 128, 5);
    k_bias_init<<<cdiv((long)N * 5, TPB), TPB, 0, stream>>>(c2b, out5, N * 5, 5);
    k_selfloop<<<cdiv((long)N * 5, TPB), TPB, 0, stream>>>(xw5, dinv, out5, N * 5, 5);
    k_edge_scatter<<<cdiv((long)E * 5, TPB), TPB, 0, stream>>>(esrc, edst, dinv, xw5, out5, E, 5);

    // ---- max-pool + softmax ----
    k_rowmax_partial<<<64, TPB, 0, stream>>>(out5, part, N);
    k_softmax_final<<<1, 32, 0, stream>>>(part, (float*)d_out, 64);
}
